// AggLineProbe_41197326303804
// MI455X (gfx1250) — matrix operand order repaired and checked by reading, not compiled, not run on hardware
//
#include <hip/hip_runtime.h>
#include <hip/hip_bf16.h>
#include <math.h>

// ---------------------------------------------------------------------------
// Problem constants (from setup_inputs): T=131072, D=2048, H=260,
// tokens_per_problem=512, tokens_per_line=16, NUM_HEADS=4, head_dim=64.
// ---------------------------------------------------------------------------
#define D_K     2048
#define H_OUT   260
#define NUMH    4
#define HD      64          // (H_OUT - NUMH) / NUMH
#define SPROB   512
#define SLINE   16
#define NLINE   (SPROB / SLINE)

// GEMM tiling
#define BM      128
#define BK      32
#define NT      17          // ceil(260/16)
#define BN      (NT * 16)   // 272 (padded with zeros)
#define NCHUNK  (D_K / BK)  // 64

typedef __attribute__((ext_vector_type(16))) __bf16 v16bf;
typedef __attribute__((ext_vector_type(8)))  __bf16 v8bf;
typedef __attribute__((ext_vector_type(8)))  float  v8f;
typedef __attribute__((ext_vector_type(4)))  int    v4i;

#define GAS __attribute__((address_space(1)))
#define LAS __attribute__((address_space(3)))

// ---------------------------------------------------------------------------
// CDNA5 async global->LDS (ASYNCcnt) with guarded fallback.
// Builtin signature (from clang diagnostic): (v4i AS1*, v4i AS3*, Ii, Ii).
// ---------------------------------------------------------------------------
#if defined(__gfx1250__) && __has_builtin(__builtin_amdgcn_global_load_async_to_lds_b128)
#define USE_ASYNC 1
#else
#define USE_ASYNC 0
#endif

__device__ __forceinline__ void cp16_g2l(void* lds, const void* g)
{
#if USE_ASYNC
    __builtin_amdgcn_global_load_async_to_lds_b128(
        (GAS v4i*)g, (LAS v4i*)lds, /*imm offset*/0, /*cpol*/0);
#else
    *(float4*)lds = *(const float4*)g;
#endif
}

__device__ __forceinline__ void wait_async0()
{
#if USE_ASYNC
#if __has_builtin(__builtin_amdgcn_s_wait_asynccnt)
    __builtin_amdgcn_s_wait_asynccnt(0);
#else
    asm volatile("s_wait_asynccnt 0x0" ::: "memory");
#endif
#endif
}

// ---------------------------------------------------------------------------
// Kernel 0 (prep): Wt[n][k] = bf16(W_down[k][n]), n padded 260 -> 272 with 0.
// Runs once per launch; 2 MB read, 1.1 MB write.
// ---------------------------------------------------------------------------
__global__ __launch_bounds__(256, 1)
void transpose_w_kernel(const float* __restrict__ Wd, __bf16* __restrict__ Wt)
{
    const int n = blockIdx.x;               // 0..271
    for (int k = threadIdx.x; k < D_K; k += 256) {
        float v = (n < H_OUT) ? Wd[(size_t)k * H_OUT + n] : 0.0f;
        Wt[(size_t)n * D_K + k] = (__bf16)v;
    }
}

// ---------------------------------------------------------------------------
// Kernel 1: act = concat(proj[:, :4], gelu(proj[:, 4:])),
//           proj = emb @ W_down + b_down via bf16 WMMA, f32 accumulate.
// Double-buffered LDS; async global->LDS DMA overlaps the WMMA compute.
// Grid: T/BM blocks of 256 threads (8 waves); wave w owns m-tile w, all 17
// n-tiles (17 x v8f accumulators).
// ---------------------------------------------------------------------------
__global__ __launch_bounds__(256, 1)
void gemm_act_kernel(const float* __restrict__ emb,
                     const __bf16* __restrict__ Wt,
                     const float* __restrict__ bd,
                     float* __restrict__ act)
{
    __shared__ __attribute__((aligned(64))) float  lAf[2][BM][BK];  // 2 x 16 KB (fp32 A tile)
    __shared__ __attribute__((aligned(64))) __bf16 lB [2][BN][BK];  // 2 x 17 KB (bf16 B tile, [N][K])

    const int tid  = threadIdx.x;
    const int wave = tid >> 5;
    const int lane = tid & 31;
    const long mbase = (long)blockIdx.x * BM;

    v8f acc[NT] = {};

    // Fragment addressing (wave32; k = 16*(lane/16)+s for A and B alike, not the ISA table order):
    //  A (16x32 MxK): M = lane%16, 16 contiguous K at kbaseA (two chunks of 8)
    //  B (32x16 KxN): N = lane%16, 16 contiguous K at kbaseB
    const int mrow   = wave * 16 + (lane & 15);
    const int kbaseA = (lane >> 4) * 16;
    const int kbaseB = (lane >> 4) * 16;

    // --- stage one K-chunk (pure 16B block copies, async when available) ---
    auto stage = [&](int k0, int buf) {
        // A: 128 rows x 32 fp32 = 1024 x 16B chunks; 4 per thread
        #pragma unroll
        for (int j = 0; j < 4; ++j) {
            const int c   = tid + j * 256;
            const int r   = c >> 3;            // row 0..127
            const int off = (c & 7) * 4;       // float offset within row
            cp16_g2l(&lAf[buf][r][off],
                     emb + (mbase + r) * (long)D_K + k0 + off);
        }
        // B: 272 rows x 32 bf16 (64B/row) = 1088 x 16B chunks; <=5 per thread
        #pragma unroll
        for (int j = 0; j < 5; ++j) {
            const int c = tid + j * 256;
            if (c < BN * 4) {
                const int n   = c >> 2;
                const int off = (c & 3) * 8;   // bf16 offset within row
                cp16_g2l(&lB[buf][n][off],
                         Wt + (size_t)n * D_K + k0 + off);
            }
        }
    };

    stage(0, 0);

    for (int kk = 0; kk < NCHUNK; ++kk) {
        const int cur = kk & 1;
        wait_async0();            // this wave's DMA for buf[cur] complete
        __syncthreads();          // all waves' tiles visible
        if (kk + 1 < NCHUNK)
            stage((kk + 1) * BK, cur ^ 1);   // DMA overlaps compute below

        // ---- A fragment: 2 x 32B fp32 LDS loads, convert to bf16 ----
        v8f va = *(const v8f*)&lAf[cur][mrow][kbaseA];
        v8f vb = *(const v8f*)&lAf[cur][mrow][kbaseA + 8];
        v16bf afrag;
        #pragma unroll
        for (int i = 0; i < 8; ++i) {
            afrag[i]     = (__bf16)va[i];
            afrag[i + 8] = (__bf16)vb[i];
        }

        // ---- 17 WMMA ops per K-chunk per wave ----
        #pragma unroll
        for (int j = 0; j < NT; ++j) {
            const int nrow = j * 16 + (lane & 15);
            v16bf bfrag = *(const v16bf*)&lB[cur][nrow][kbaseB];
            acc[j] = __builtin_amdgcn_wmma_f32_16x16x32_bf16(
                         false, afrag, false, bfrag,
                         (short)0, acc[j], false, false);
        }
        __syncthreads();          // done reading buf[cur] before it is refilled
    }

    // ---- epilogue: +b_down, exact-erf GELU on cols >= 4, store act ----
    const int col   = lane & 15;
    const int rbase = (lane >> 4) * 8;
    const long row0 = mbase + wave * 16;
    #pragma unroll
    for (int j = 0; j < NT; ++j) {
        const int gcol = j * 16 + col;
        if (gcol < H_OUT) {
            const float b = bd[gcol];
            #pragma unroll
            for (int r = 0; r < 8; ++r) {
                float x = acc[j][r] + b;
                if (gcol >= NUMH)
                    x = 0.5f * x * (1.0f + erff(x * 0.70710678118654752f));
                act[(size_t)(row0 + rbase + r) * H_OUT + gcol] = x;
            }
        }
    }
}

// ---------------------------------------------------------------------------
// Kernel 2: pooling + logits. One block (256 threads) per 512-token problem.
// Linearity: logit = b + sum_h sum_s softmax(a[:,h])[s] * hdot[s][h],
// hdot[s][h] = <act[s, 4+64h : 4+64h+64], W_final[64h : 64h+64]>.
// ---------------------------------------------------------------------------
__global__ __launch_bounds__(256, 1)
void pool_kernel(const float* __restrict__ act,
                 const float* __restrict__ Wf,
                 const float* __restrict__ bfin_p,
                 const int*   __restrict__ labels,
                 float* __restrict__ tokL,
                 float* __restrict__ lineL,
                 int*   __restrict__ lineLab,
                 float* __restrict__ probL,
                 int*   __restrict__ probLab)
{
    __shared__ float wf[H_OUT - NUMH];
    __shared__ float tA[SPROB][NUMH];
    __shared__ float tH[SPROB][NUMH];
    __shared__ float headAcc[NUMH];

    const int tid = threadIdx.x;
    const int p   = blockIdx.x;
    const float bfin = bfin_p[0];

    wf[tid] = Wf[tid];
    __syncthreads();

    for (int lt = tid; lt < SPROB; lt += 256) {
        const size_t trow = ((size_t)p * SPROB + lt);
        const float* row  = act + trow * H_OUT;
        #pragma unroll
        for (int h = 0; h < NUMH; ++h) tA[lt][h] = row[h];

        const float4* v4 = (const float4*)(row + NUMH);   // 16B aligned
        const float4* w4 = (const float4*)wf;
        float hd[NUMH];
        #pragma unroll
        for (int h = 0; h < NUMH; ++h) {
            float s = 0.f;
            #pragma unroll 4
            for (int q = 0; q < HD / 4; ++q) {
                float4 a = v4[h * (HD / 4) + q];
                float4 w = w4[h * (HD / 4) + q];
                s += a.x * w.x + a.y * w.y + a.z * w.z + a.w * w.w;
            }
            hd[h] = s;
            tH[lt][h] = s;
        }
        tokL[trow] = hd[0] + hd[1] + hd[2] + hd[3] + bfin;
    }
    __syncthreads();

    if (tid < NLINE) {
        const int li = tid;
        float logit = bfin;
        #pragma unroll
        for (int h = 0; h < NUMH; ++h) {
            float m = -3.4e38f;
            for (int s = 0; s < SLINE; ++s)
                m = fmaxf(m, tA[li * SLINE + s][h]);
            float den = 0.f, num = 0.f;
            for (int s = 0; s < SLINE; ++s) {
                float e = __expf(tA[li * SLINE + s][h] - m);
                den += e;
                num += e * tH[li * SLINE + s][h];
            }
            logit += num / den;
        }
        lineL[(size_t)p * NLINE + li] = logit;

        int lab = labels[(size_t)p * SPROB + li * SLINE];
        for (int s = 1; s < SLINE; ++s)
            lab = min(lab, labels[(size_t)p * SPROB + li * SLINE + s]);
        lineLab[(size_t)p * NLINE + li] = lab;
    }

    if (tid < NUMH) {
        const int h = tid;
        float m = -3.4e38f;
        for (int s = 0; s < SPROB; ++s) m = fmaxf(m, tA[s][h]);
        float den = 0.f, num = 0.f;
        for (int s = 0; s < SPROB; ++s) {
            float e = __expf(tA[s][h] - m);
            den += e;
            num += e * tH[s][h];
        }
        headAcc[h] = num / den;
    }
    if (tid == NUMH) {
        int lab = labels[(size_t)p * SPROB];
        for (int s = 1; s < SPROB; ++s)
            lab = min(lab, labels[(size_t)p * SPROB + s]);
        probLab[p] = lab;
    }
    __syncthreads();
    if (tid == 0)
        probL[p] = headAcc[0] + headAcc[1] + headAcc[2] + headAcc[3] + bfin;
}

// ---------------------------------------------------------------------------
// Launch
// ---------------------------------------------------------------------------
extern "C" void kernel_launch(void* const* d_in, const int* in_sizes, int n_in,
                              void* d_out, int out_size, void* d_ws, size_t ws_size,
                              hipStream_t stream)
{
    const float* emb    = (const float*)d_in[0];
    const float* Wd     = (const float*)d_in[1];
    const float* bd     = (const float*)d_in[2];
    const float* Wf     = (const float*)d_in[3];
    const float* bfinal = (const float*)d_in[4];
    const int*   labels = (const int*)d_in[5];

    const int  H = in_sizes[2];                 // 260
    const int  D = in_sizes[1] / H;             // 2048
    const long T = (long)in_sizes[0] / D;       // 131072
    const int  P = (int)(T / SPROB);            // 256
    const int  L = (int)(T / SLINE);            // 8192

    // workspace layout: act [T x H] fp32, then Wt [272 x 2048] bf16
    float*  act = (float*)d_ws;
    size_t  actBytes = ((size_t)T * H * sizeof(float) + 255) & ~(size_t)255;
    __bf16* Wt  = (__bf16*)((char*)d_ws + actBytes);

    float* out     = (float*)d_out;
    float* tokL    = out;                            // [T]
    float* lineL   = out + T;                        // [L]
    int*   lineLab = (int*)(out + T + L);            // [L] (int32 bits)
    float* probL   = out + T + (long)L * 2;          // [P]
    int*   probLab = (int*)(out + T + (long)L * 2 + P); // [P]

    transpose_w_kernel<<<dim3(BN), 256, 0, stream>>>(Wd, Wt);

    gemm_act_kernel<<<dim3((unsigned)(T / BM)), 256, 0, stream>>>(emb, Wt, bd, act);

    pool_kernel<<<dim3((unsigned)P), 256, 0, stream>>>(
        act, Wf, bfinal, labels, tokL, lineL, lineLab, probL, probLab);
}
